// ParaLIF_41059887349804
// MI455X (gfx1250) — matrix operand order repaired and checked by reading, not compiled, not run on hardware
//
#include <hip/hip_runtime.h>

// ParaLIF: X = inputs @ W^T + b  (WMMA f16, f32 accum), then the FFT
// double-exponential filter re-expressed as an exact 2-state IIR scan:
//   syn[t] = a*syn[t-1] + (1-a)*X[t] ;  mem[t] = a*mem[t-1] + syn[t]

#define B_  16
#define T_  4096
#define DI  512
#define DH  512

#define TM  64     // workgroup M tile (time rows)
#define TN  256    // workgroup N tile (hidden cols)
#define TK  32     // K slab per WMMA step
#define NK  (DI / TK)

#define ALPHA_F      0.36787944117144233f   // exp(-1)
#define ONE_M_ALPHA  0.63212055882855767f   // 1 - exp(-1)

typedef __attribute__((ext_vector_type(16))) _Float16     v16h;
typedef __attribute__((ext_vector_type(8)))  float        v8f;
typedef __attribute__((ext_vector_type(4)))  unsigned int u32x4;

union FragH { unsigned int u[8]; u32x4 q[2]; v16h h; };

__device__ __forceinline__ unsigned int pack2_f16(float a, float b) {
  union { _Float16 h[2]; unsigned int u; } p;
  p.h[0] = (_Float16)a;
  p.h[1] = (_Float16)b;
  return p.u;
}

// ---------------------------------------------------------------------------
// GEMM: X[b,t,h] = sum_k inputs[b,t,k] * W[h,k] + bias[h]
// Grid: (DH/TN, T_/TM, B_), block 256 (8 waves). Wave computes 32x64 output.
// LDS layouts are "fragment major" so every WMMA operand fragment is two
// contiguous ds_load_b128 per lane (no register shuffles).
// ---------------------------------------------------------------------------
__global__ __launch_bounds__(256, 2)
void paralif_gemm_f16_wmma(const float* __restrict__ inp,
                           const float* __restrict__ W,
                           const float* __restrict__ bias,
                           float* __restrict__ X)
{
  // f16 K-pairs (dword = {k_even, k_odd}); both arrays keep a row/col's 16
  // K-pairs contiguous (64B) -> fragment loads are contiguous b128s.
  __shared__ unsigned int lds_a[TM * (TK / 2)];   // [row][k2]   4 KB
  __shared__ unsigned int lds_b[TN * (TK / 2)];   // [col][k2]  16 KB

  const int tid  = threadIdx.x;
  const int lane = tid & 31;
  const int wave = tid >> 5;        // 0..7
  const int wm   = wave & 1;        // M half: rows [0,32) or [32,64)
  const int wn   = wave >> 1;       // N quarter: 64 cols each
  const int l16  = lane & 15;
  const int hi   = lane >> 4;       // 0 = lanes 0-15, 1 = lanes 16-31

  const int h0 = blockIdx.x * TN;
  const int t0 = blockIdx.y * TM;
  const int bb = blockIdx.z;

  // Cooperative-fill mapping: A: thread -> (row, 8-float chunk); B: thread = col
  const int arow = tid >> 2;            // 0..63
  const int acol = (tid & 3) * 8;       // 0,8,16,24
  const float* gArow = inp + ((size_t)bb * T_ + (size_t)(t0 + arow)) * DI + acol;
  const float* gWrow = W + (size_t)(h0 + tid) * DI;

  v8f acc[2][4];
  v8f zero = {};
#pragma unroll
  for (int mt = 0; mt < 2; ++mt)
#pragma unroll
    for (int nt = 0; nt < 4; ++nt) acc[mt][nt] = zero;

  for (int ks = 0; ks < NK; ++ks) {
    const int k0 = ks * TK;

    // ---- fill LDS (fp32 global -> f16 pairs) ----
    {
      const float* ga = gArow + k0;
      float4 f0 = *(const float4*)ga;
      float4 f1 = *(const float4*)(ga + 4);
      u32x4 apk = { pack2_f16(f0.x, f0.y), pack2_f16(f0.z, f0.w),
                    pack2_f16(f1.x, f1.y), pack2_f16(f1.z, f1.w) };
      *(u32x4*)&lds_a[arow * (TK / 2) + (acol >> 1)] = apk;

      const float* gw = gWrow + k0;
      unsigned int* lb = &lds_b[tid * (TK / 2)];
#pragma unroll
      for (int q = 0; q < 4; ++q) {
        float4 w0 = *(const float4*)(gw + q * 8);
        float4 w1 = *(const float4*)(gw + q * 8 + 4);
        u32x4 bpk = { pack2_f16(w0.x, w0.y), pack2_f16(w0.z, w0.w),
                      pack2_f16(w1.x, w1.y), pack2_f16(w1.z, w1.w) };
        *(u32x4*)&lb[q * 4] = bpk;
      }
      if (ks + 1 < NK) {                    // warm L2/WGP$ for next K slab
        __builtin_prefetch(ga + TK, 0, 0);
        __builtin_prefetch(gw + TK, 0, 0);
      }
    }
    __syncthreads();

    // ---- A fragments: lane<16 holds K {0..7,16..23}, lane>=16 {8..15,24..31}
    FragH a[2];
#pragma unroll
    for (int mt = 0; mt < 2; ++mt) {
      const unsigned int* pa =
          &lds_a[(wm * 32 + mt * 16 + l16) * (TK / 2) + hi * 4];
      a[mt].q[0] = *(const u32x4*)pa;        // k2 {0..3} / {4..7}
      a[mt].q[1] = *(const u32x4*)(pa + 8);  // k2 {8..11} / {12..15}
    }

    // ---- B fragments (contiguous b128 pair) + WMMA
#pragma unroll
    for (int nt = 0; nt < 4; ++nt) {
      FragH bf;
      const unsigned int* pb =
          &lds_b[(wn * 64 + nt * 16 + l16) * (TK / 2) + hi * 4];
      bf.q[0] = *(const u32x4*)pb;           // k2 {0..3} / {4..7}
      bf.q[1] = *(const u32x4*)(pb + 8);     // k2 {8..11} / {12..15}
#pragma unroll
      for (int mt = 0; mt < 2; ++mt) {
        acc[mt][nt] = __builtin_amdgcn_wmma_f32_16x16x32_f16(
            false, a[mt].h, false, bf.h, (short)0, acc[mt][nt], false, false);
      }
    }
    __syncthreads();
  }

  // ---- epilogue: add bias, store fp32 X; C/D layout: VGPR r -> M = r + 8*hi
#pragma unroll
  for (int nt = 0; nt < 4; ++nt) {
    const int h = h0 + wn * 64 + nt * 16 + l16;
    const float bv = bias[h];
#pragma unroll
    for (int mt = 0; mt < 2; ++mt) {
      const int trow = t0 + wm * 32 + mt * 16 + hi * 8;
      float* po = X + ((size_t)bb * T_ + trow) * DH + h;
#pragma unroll
      for (int r = 0; r < 8; ++r)
        po[(size_t)r * DH] = acc[mt][nt][r] + bv;
    }
  }
}

// ---------------------------------------------------------------------------
// Scan: exact replacement for the FFT double-exponential filter.
// One thread per (b,h); coalesced along h. Grid (DH/256, B_), block 256.
// ---------------------------------------------------------------------------
__global__ __launch_bounds__(256)
void paralif_scan(const float* __restrict__ X, float* __restrict__ out)
{
  const int h  = blockIdx.x * 256 + threadIdx.x;
  const int bb = blockIdx.y;
  float syn = 0.0f, mem = 0.0f;
  size_t idx = (size_t)bb * T_ * DH + h;
  for (int t = 0; t < T_; ++t, idx += DH) {
    const float x = X[idx];
    syn = __builtin_fmaf(ALPHA_F, syn, ONE_M_ALPHA * x);  // k-filter
    mem = __builtin_fmaf(ALPHA_F, mem, syn);              // l-filter
    out[idx] = mem;
  }
}

// ---------------------------------------------------------------------------
extern "C" void kernel_launch(void* const* d_in, const int* in_sizes, int n_in,
                              void* d_out, int out_size, void* d_ws, size_t ws_size,
                              hipStream_t stream)
{
  (void)in_sizes; (void)n_in; (void)out_size; (void)ws_size;
  const float* inp  = (const float*)d_in[0];
  const float* W    = (const float*)d_in[1];
  const float* bias = (const float*)d_in[2];
  float* out = (float*)d_out;
  float* X   = (float*)d_ws;   // B_*T_*DH*4 = 128 MiB scratch for pre-scan X

  dim3 ggrid(DH / TN, T_ / TM, B_);         // (2, 64, 16) = 2048 workgroups
  paralif_gemm_f16_wmma<<<ggrid, 256, 0, stream>>>(inp, W, bias, X);

  dim3 sgrid(DH / 256, B_);                 // 8192 independent scans
  paralif_scan<<<sgrid, 256, 0, stream>>>(X, out);
}
